// SubGnnNp_35227321762431
// MI455X (gfx1250) — matrix operand order repaired and checked by reading, not compiled, not run on hardware
//
#include <hip/hip_runtime.h>
#include <hip/hip_bf16.h>

typedef __bf16 bf16;
typedef bf16  v16bf __attribute__((ext_vector_type(16)));
typedef float v8f   __attribute__((ext_vector_type(8)));

#define NN 64
#define DD 128

// ---- bf16 weight arena layout (element offsets inside d_ws) ----
#define OFF_EM_FC1 0                       // conv.edge_merger.fc1.w [128 x 384]
#define OFF_EM_FC2 (128*384)               // conv.edge_merger.fc2.w [128 x 128]
#define OFF_MLP    (OFF_EM_FC2 + 128*128)  // mlps: l1_0,l2_0,l1_1,l2_1,l1_2,l2_2, each [128x128]
#define OFF_WK     (OFF_MLP + 6*128*128)   // attn[i].wk, i=0..3
#define OFF_WV     (OFF_WK + 4*128*128)    // attn[i].wv, i=0..3
#define WBF_TOTAL  (OFF_WV + 4*128*128)    // 294912 elements

#define WMMA_BF16(a, b, c) \
  __builtin_amdgcn_wmma_f32_16x16x32_bf16(false, (a), false, (b), (short)0, (c), false, false)

// ---------------- WMMA fragment loaders (contiguous k = 16*(lane/16) + e in both A and B) ----------------
// A 16x32 bf16: lane holds row M=lane%16; K = e + ((lane>>4)<<4)
__device__ __forceinline__ v16bf frag_a(const bf16* base, int row, int ld, int k0, int lane) {
  const bf16* p = base + row * ld + k0 + ((lane >> 4) << 4);
  v16bf f;
#pragma unroll
  for (int e = 0; e < 16; ++e) f[e] = p[e];
  return f;
}

// B 32x16 bf16 built from row-major W[out][in] (B[k][n] = W[n][k]):
// lane holds column N=lane%16; K = e + ((lane>>4)<<4)  -> 16 contiguous elements of W row N.
__device__ __forceinline__ v16bf frag_b_w(const bf16* W, int nrow, int ld, int k0, int lane) {
  const bf16* p = W + nrow * ld + k0 + ((lane >> 4) << 4);
  v16bf f;
#pragma unroll
  for (int e = 0; e < 16; ++e) f[e] = p[e];
  return f;
}

// B 32x16 bf16 from row-major S[K][Ncols] (true B layout): column-strided loads.
__device__ __forceinline__ v16bf frag_b_col(const bf16* S, int col, int ld, int k0, int lane) {
  const bf16* p = S + (k0 + ((lane >> 4) << 4)) * ld + col;
  v16bf f;
#pragma unroll
  for (int e = 0; e < 16; ++e) f[e] = p[e * ld];
  return f;
}

// C(64x128)+= A(64xK, LDS row-major) * W^T, W bf16 row-major [128 x ldw], k-offset kw0 into W rows
__device__ __forceinline__ void gemm_w(const bf16* A, int lda, const bf16* W, int ldw, int kw0,
                                       int K, int lane, int rb, int cb, v8f acc[4]) {
  const int l15 = lane & 15;
  for (int kk = 0; kk < K; kk += 32) {
    v16bf a = frag_a(A, rb + l15, lda, kk, lane);
#pragma unroll
    for (int t = 0; t < 4; ++t) {
      v16bf b = frag_b_w(W, cb + t * 16 + l15, ldw, kw0 + kk, lane);
      acc[t] = WMMA_BF16(a, b, acc[t]);
    }
  }
}

// C(64x128)+= A(64xK) * S, S bf16 row-major [K x ldsb] in LDS
__device__ __forceinline__ void gemm_s(const bf16* A, int lda, const bf16* S, int ldsb,
                                       int K, int lane, int rb, int cb, v8f acc[4]) {
  const int l15 = lane & 15;
  for (int kk = 0; kk < K; kk += 32) {
    v16bf a = frag_a(A, rb + l15, lda, kk, lane);
#pragma unroll
    for (int t = 0; t < 4; ++t) {
      v16bf b = frag_b_col(S, cb + t * 16 + l15, ldsb, kk, lane);
      acc[t] = WMMA_BF16(a, b, acc[t]);
    }
  }
}

// D layout: VGPR r of lane L -> row = r + (L>=16 ? 8 : 0), col = L%16
__device__ __forceinline__ void store_acc(bf16* D_, int ldd, int lane, int rb, int cb,
                                          const v8f acc[4], const float* bias, bool relu) {
  const int l15 = lane & 15;
  const int r0 = rb + ((lane >> 4) << 3);
#pragma unroll
  for (int t = 0; t < 4; ++t) {
    int col = cb + t * 16 + l15;
    float bv = bias ? bias[col] : 0.0f;
#pragma unroll
    for (int r = 0; r < 8; ++r) {
      float v = acc[t][r] + bv;
      if (relu) v = fmaxf(v, 0.0f);
      D_[(r0 + r) * ldd + col] = (bf16)v;
    }
  }
}

// ---------------- weight f32 -> bf16 conversion ----------------
struct WConvArgs {
  const float* src[16];
  int n[16];
  int off[16];
};

__global__ __launch_bounds__(256) void wconvert_kernel(WConvArgs A, bf16* dst) {
  const int gsz = gridDim.x * blockDim.x;
  const int gid = blockIdx.x * blockDim.x + threadIdx.x;
  for (int t = 0; t < 16; ++t) {
    const float* s = A.src[t];
    bf16* d = dst + A.off[t];
    for (int i = gid; i < A.n[t]; i += gsz) d[i] = (bf16)s[i];
  }
}

// ---------------- fused per-batch-element temporal conv ----------------
struct TemConvParams {
  const int* idx; const float* cut;
  const int* nid; const int* eid; const float* ets;
  const float* n2n; const float* e2n;
  const float* node_emb; const float* edge_emb;
  const float* tfreq; const float* tphase;
  const bf16* wbf;
  const float* em1b; const float* em2b;
  const float* l1b[3]; const float* l2b[3];
  const float* lng[3]; const float* lnb[3];
  const float* wq[4]; const float* fcw[4]; const float* fcb[4];
  const float* alng[4]; const float* alnb[4];
  const float* m1w[4]; const float* m1b[4];
  const float* m2w[4]; const float* m2b[4];
  const float* transw; const float* queryw;
  const float* flng; const float* flnb;
  float* out; // [B,128]
};

__global__ __launch_bounds__(256) void tem_conv_kernel(TemConvParams P) {
  const int b    = blockIdx.x;
  const int tid  = threadIdx.x;
  const int lane = tid & 31;
  const int wv   = tid >> 5;
  const int rb   = (wv >> 1) * 16;
  const int cb   = (wv & 1) * 64;

  __shared__ bf16 sB[NN * DD];   // nfeat -> efeat halves -> relu(fc1) -> hn buffers
  __shared__ bf16 sC[NN * DD];   // node_efeat half -> mlp tmp -> k/v
  __shared__ bf16 sH[NN * DD];   // running h
  __shared__ bf16 sM[NN * NN];   // e2n / n2n (bf16)
  __shared__ float sSrc[DD], sQ[DD], sVec[DD], sO[DD];
  __shared__ float sLoc[4 * DD];
  __shared__ float sTx[4 * DD];
  __shared__ float sLog[NN], sStatM[NN], sStatR[NN];
  __shared__ int   sNid[NN];
  __shared__ float sRed[8];

  // ---- gather nfeat (bf16), src feature, nid, e2n ----
  for (int i = tid; i < NN * DD; i += 256) {
    int n = i >> 7, d = i & 127;
    sB[i] = (bf16)P.node_emb[(long)P.nid[b * NN + n] * DD + d];
  }
  if (tid < DD) sSrc[tid] = P.node_emb[(long)P.idx[b] * DD + tid];
  if (tid < NN) sNid[tid] = P.nid[b * NN + tid];
  for (int i = tid; i < NN * NN; i += 256) sM[i] = (bf16)P.e2n[(long)b * NN * NN + i];
  __syncthreads();

  // ---- edge_merger fc1, accumulated over 3 K-chunks of 128 ----
  v8f fc1acc[4] = {};
  gemm_w(sB, DD, P.wbf + OFF_EM_FC1, 384, 0, DD, lane, rb, cb, fc1acc);  // chunk0: nfeat
  __syncthreads();

  const float cut = P.cut[b];
  // half0: t_emb = cos(dt*freq + phase)
  for (int i = tid; i < NN * DD; i += 256) {
    int n = i >> 7, d = i & 127;
    float dt = cut - P.ets[b * NN + n];
    sB[i] = (bf16)cosf(dt * P.tfreq[d] + P.tphase[d]);
  }
  __syncthreads();
  {
    v8f acc[4] = {};
    gemm_s(sM, NN, sB, DD, NN, lane, rb, cb, acc);       // node_efeat half0 = e2n @ t_emb
    store_acc(sC, DD, lane, rb, cb, acc, nullptr, false);
  }
  __syncthreads();
  gemm_w(sC, DD, P.wbf + OFF_EM_FC1, 384, 128, DD, lane, rb, cb, fc1acc); // chunk1
  __syncthreads();

  // half1: edge embeddings
  for (int i = tid; i < NN * DD; i += 256) {
    int n = i >> 7, d = i & 127;
    sB[i] = (bf16)P.edge_emb[(long)P.eid[b * NN + n] * DD + d];
  }
  __syncthreads();
  {
    v8f acc[4] = {};
    gemm_s(sM, NN, sB, DD, NN, lane, rb, cb, acc);       // node_efeat half1
    store_acc(sC, DD, lane, rb, cb, acc, nullptr, false);
  }
  __syncthreads();
  gemm_w(sC, DD, P.wbf + OFF_EM_FC1, 384, 256, DD, lane, rb, cb, fc1acc); // chunk2
  __syncthreads();
  store_acc(sB, DD, lane, rb, cb, fc1acc, P.em1b, true); // relu(fc1) -> sB
  __syncthreads();
  {
    v8f acc[4] = {};
    gemm_w(sB, DD, P.wbf + OFF_EM_FC2, DD, 0, DD, lane, rb, cb, acc);
    store_acc(sH, DD, lane, rb, cb, acc, P.em2b, false); // h = feats[0]
  }
  __syncthreads();

  // ---- 4 features: attention fused after each propagation step ----
  for (int f = 0; f < 4; ++f) {
    // k = h @ wk^T -> sC
    {
      v8f acc[4] = {};
      gemm_w(sH, DD, P.wbf + OFF_WK + f * DD * DD, DD, 0, DD, lane, rb, cb, acc);
      store_acc(sC, DD, lane, rb, cb, acc, nullptr, false);
    }
    if (tid < DD) {  // q = src @ wq^T (f32 VALU)
      const float* w = P.wq[f] + tid * DD;
      float s = 0.f;
      for (int j = 0; j < DD; ++j) s += sSrc[j] * w[j];
      sQ[tid] = s;
    }
    __syncthreads();
    if (tid < NN) {  // logits + mask
      float s = 0.f;
      for (int j = 0; j < DD; ++j) s += sQ[j] * (float)sC[tid * DD + j];
      s *= 0.08838834764831845f;  // 1/sqrt(128)
      if (sNid[tid] == 0) s = -1e10f;
      sLog[tid] = s;
    }
    __syncthreads();
    if (tid == 0) {  // softmax over 64
      float mx = -1e30f;
      for (int n = 0; n < NN; ++n) mx = fmaxf(mx, sLog[n]);
      float sum = 0.f;
      for (int n = 0; n < NN; ++n) { float e = __expf(sLog[n] - mx); sLog[n] = e; sum += e; }
      sRed[0] = 1.0f / sum;
    }
    __syncthreads();
    const float rsum = sRed[0];
    {  // v = h @ wv^T -> sC (reuse)
      v8f acc[4] = {};
      gemm_w(sH, DD, P.wbf + OFF_WV + f * DD * DD, DD, 0, DD, lane, rb, cb, acc);
      store_acc(sC, DD, lane, rb, cb, acc, nullptr, false);
    }
    __syncthreads();
    if (tid < DD) {  // out = attn @ v
      float s = 0.f;
      for (int n = 0; n < NN; ++n) s += sLog[n] * (float)sC[n * DD + tid];
      sVec[tid] = s * rsum;
    }
    __syncthreads();
    if (tid < DD) {  // fc(out) + residual(src)
      const float* w = P.fcw[f] + tid * DD;
      float s = P.fcb[f][tid];
      for (int j = 0; j < DD; ++j) s += w[j] * sVec[j];
      sO[tid] = s + sSrc[tid];
    }
    __syncthreads();
    if (tid == 0) {  // LN stats
      float m = 0.f;
      for (int j = 0; j < DD; ++j) m += sO[j];
      m /= DD;
      float v2 = 0.f;
      for (int j = 0; j < DD; ++j) { float d = sO[j] - m; v2 += d * d; }
      sRed[0] = m; sRed[1] = rsqrtf(v2 / DD + 1e-5f);
    }
    __syncthreads();
    if (tid < DD) sO[tid] = (sO[tid] - sRed[0]) * sRed[1] * P.alng[f][tid] + P.alnb[f][tid];
    __syncthreads();
    if (tid < DD) {  // merger fc1(cat(out, src)) + relu
      const float* w = P.m1w[f] + tid * 256;
      float s = P.m1b[f][tid];
      for (int j = 0; j < DD; ++j) s += w[j] * sO[j];
      for (int j = 0; j < DD; ++j) s += w[DD + j] * sSrc[j];
      sVec[tid] = fmaxf(s, 0.0f);
    }
    __syncthreads();
    if (tid < DD) {  // merger fc2 -> loc[f]
      const float* w = P.m2w[f] + tid * DD;
      float s = P.m2b[f][tid];
      for (int j = 0; j < DD; ++j) s += w[j] * sVec[j];
      sLoc[f * DD + tid] = s;
    }
    __syncthreads();

    if (f < 3) {  // propagation step f
      for (int i = tid; i < NN * NN; i += 256) sM[i] = (bf16)P.n2n[(long)b * NN * NN + i];
      __syncthreads();
      {  // hn0 = n2n @ h
        v8f acc[4] = {};
        gemm_s(sM, NN, sH, DD, NN, lane, rb, cb, acc);
        store_acc(sB, DD, lane, rb, cb, acc, nullptr, false);
      }
      __syncthreads();
      {  // t = relu(l1(hn0))
        v8f acc[4] = {};
        gemm_w(sB, DD, P.wbf + OFF_MLP + (2 * f) * DD * DD, DD, 0, DD, lane, rb, cb, acc);
        store_acc(sC, DD, lane, rb, cb, acc, P.l1b[f], true);
      }
      __syncthreads();
      {  // hn2 = l2(t)
        v8f acc[4] = {};
        gemm_w(sC, DD, P.wbf + OFF_MLP + (2 * f + 1) * DD * DD, DD, 0, DD, lane, rb, cb, acc);
        store_acc(sB, DD, lane, rb, cb, acc, P.l2b[f], false);
      }
      __syncthreads();
      if (tid < NN) {  // per-row LN stats
        float m = 0.f;
        for (int j = 0; j < DD; ++j) m += (float)sB[tid * DD + j];
        m /= DD;
        float v2 = 0.f;
        for (int j = 0; j < DD; ++j) { float d = (float)sB[tid * DD + j] - m; v2 += d * d; }
        sStatM[tid] = m; sStatR[tid] = rsqrtf(v2 / DD + 1e-5f);
      }
      __syncthreads();
      for (int i = tid; i < NN * DD; i += 256) {  // relu(LN) + blend
        int n = i >> 7, d = i & 127;
        float x = ((float)sB[i] - sStatM[n]) * sStatR[n] * P.lng[f][d] + P.lnb[f][d];
        x = fmaxf(x, 0.0f);
        sH[i] = (bf16)(0.2f * (float)sH[i] + 0.8f * x);
      }
      __syncthreads();
    }
  }

  // ---- fusion: tx = tanh(trans_w @ xflat); softmax(tx_k . query_w); LN(sum w_k loc_k) ----
  for (int o = tid; o < 4 * DD; o += 256) {
    const float* tw = P.transw + o * (4 * DD);
    float s = 0.f;
    for (int j = 0; j < 4 * DD; ++j) s += tw[j] * sLoc[j];
    sTx[o] = tanhf(s);
  }
  __syncthreads();
  if (tid == 0) {
    float wgt[4]; float mx = -1e30f;
    for (int k = 0; k < 4; ++k) {
      float s = 0.f;
      for (int j = 0; j < DD; ++j) s += sTx[k * DD + j] * P.queryw[j];
      wgt[k] = s; mx = fmaxf(mx, s);
    }
    float sum = 0.f;
    for (int k = 0; k < 4; ++k) { wgt[k] = __expf(wgt[k] - mx); sum += wgt[k]; }
    for (int k = 0; k < 4; ++k) sRed[k] = wgt[k] / sum;
  }
  __syncthreads();
  if (tid < DD) {
    float s = 0.f;
    for (int k = 0; k < 4; ++k) s += sRed[k] * sLoc[k * DD + tid];
    sVec[tid] = s;
  }
  __syncthreads();
  if (tid == 0) {
    float m = 0.f;
    for (int j = 0; j < DD; ++j) m += sVec[j];
    m /= DD;
    float v2 = 0.f;
    for (int j = 0; j < DD; ++j) { float d = sVec[j] - m; v2 += d * d; }
    sRed[0] = m; sRed[1] = rsqrtf(v2 / DD + 1e-5f);
  }
  __syncthreads();
  if (tid < DD)
    P.out[(long)b * DD + tid] = (sVec[tid] - sRed[0]) * sRed[1] * P.flng[tid] + P.flnb[tid];
}

// ---------------- affinity head + sigmoid ----------------
__global__ __launch_bounds__(128) void affinity_kernel(const float* se, const float* te,
                                                       const float* w1, const float* b1,
                                                       const float* w2, const float* b2,
                                                       float* out) {
  const int b = blockIdx.x, tid = threadIdx.x;
  __shared__ float sh[128];
  const float* s = se + (long)b * 128;
  const float* t = te + (long)b * 128;
  const float* w = w1 + tid * 256;
  float acc = b1[tid];
  for (int j = 0; j < 128; ++j) acc += w[j] * s[j];
  for (int j = 0; j < 128; ++j) acc += w[128 + j] * t[j];
  sh[tid] = fmaxf(acc, 0.0f) * w2[tid];
  __syncthreads();
  if (tid < 64) sh[tid] += sh[tid + 64];
  __syncthreads();
  if (tid < 32) sh[tid] += sh[tid + 32];
  __syncthreads();
  if (tid == 0) {
    float r = b2[0];
    for (int j = 0; j < 32; ++j) r += sh[j];
    out[b] = 1.0f / (1.0f + __expf(-r));
  }
}

// ---------------- host ----------------
// Input index map: 0..12 raw tensors (setup_inputs insertion order), then params
// flattened jax-pytree style (dict keys sorted, lists in order), base 13:
//  13 aff.fc1.b 14 aff.fc1.w 15 aff.fc2.b 16 aff.fc2.w
//  17+11i: attn[i]: fc.b, fc.w, ln.b, ln.g, m.fc1.b, m.fc1.w, m.fc2.b, m.fc2.w, wk, wq, wv
//  61 em.fc1.b 62 em.fc1.w 63 em.fc2.b 64 em.fc2.w
//  65+2i: lns[i].b, lns[i].g
//  71+4i: mlp[i].l1.b, l1.w, l2.b, l2.w
//  83 edge_emb  84 fus.ln.b 85 fus.ln.g 86 query_w 87 trans_w
//  88 node_emb  89 time_freq 90 time_phase
extern "C" void kernel_launch(void* const* d_in, const int* in_sizes, int n_in,
                              void* d_out, int out_size, void* d_ws, size_t ws_size,
                              hipStream_t stream) {
  (void)in_sizes; (void)n_in; (void)out_size; (void)ws_size;
  auto F = [&](int i) { return (const float*)d_in[i]; };
  auto I = [&](int i) { return (const int*)d_in[i]; };

  bf16* wbf    = (bf16*)d_ws;
  float* src_e = (float*)((char*)d_ws + (1u << 20));
  float* tgt_e = (float*)((char*)d_ws + (1u << 20) + 4096u * 128u * 4u);

  // 1) weights -> bf16
  WConvArgs ca;
  ca.src[0] = F(62); ca.n[0] = 128 * 384; ca.off[0] = OFF_EM_FC1;
  ca.src[1] = F(64); ca.n[1] = 128 * 128; ca.off[1] = OFF_EM_FC2;
  for (int i = 0; i < 3; ++i) {
    ca.src[2 + 2 * i] = F(71 + 4 * i + 1); ca.n[2 + 2 * i] = 128 * 128;
    ca.off[2 + 2 * i] = OFF_MLP + (2 * i) * 128 * 128;
    ca.src[3 + 2 * i] = F(71 + 4 * i + 3); ca.n[3 + 2 * i] = 128 * 128;
    ca.off[3 + 2 * i] = OFF_MLP + (2 * i + 1) * 128 * 128;
  }
  for (int i = 0; i < 4; ++i) {
    ca.src[8 + i]  = F(17 + 11 * i + 8);  ca.n[8 + i]  = 128 * 128; ca.off[8 + i]  = OFF_WK + i * 128 * 128;
    ca.src[12 + i] = F(17 + 11 * i + 10); ca.n[12 + i] = 128 * 128; ca.off[12 + i] = OFF_WV + i * 128 * 128;
  }
  hipLaunchKernelGGL(wconvert_kernel, dim3(256), dim3(256), 0, stream, ca, wbf);

  // 2) shared params
  TemConvParams P{};
  P.node_emb = F(88); P.edge_emb = F(83);
  P.tfreq = F(89); P.tphase = F(90);
  P.wbf = wbf;
  P.em1b = F(61); P.em2b = F(63);
  for (int i = 0; i < 3; ++i) {
    P.l1b[i] = F(71 + 4 * i + 0);
    P.l2b[i] = F(71 + 4 * i + 2);
    P.lnb[i] = F(65 + 2 * i + 0);
    P.lng[i] = F(65 + 2 * i + 1);
  }
  for (int i = 0; i < 4; ++i) {
    int a = 17 + 11 * i;
    P.fcb[i] = F(a + 0); P.fcw[i] = F(a + 1);
    P.alnb[i] = F(a + 2); P.alng[i] = F(a + 3);
    P.m1b[i] = F(a + 4); P.m1w[i] = F(a + 5);
    P.m2b[i] = F(a + 6); P.m2w[i] = F(a + 7);
    P.wq[i]  = F(a + 9);
  }
  P.flnb = F(84); P.flng = F(85); P.queryw = F(86); P.transw = F(87);
  P.cut = F(2);

  // src branch
  TemConvParams Ps = P;
  Ps.idx = I(0); Ps.nid = I(3); Ps.eid = I(4); Ps.ets = F(5);
  Ps.n2n = F(6); Ps.e2n = F(7); Ps.out = src_e;
  hipLaunchKernelGGL(tem_conv_kernel, dim3(4096), dim3(256), 0, stream, Ps);

  // tgt branch
  TemConvParams Pt = P;
  Pt.idx = I(1); Pt.nid = I(8); Pt.eid = I(9); Pt.ets = F(10);
  Pt.n2n = F(11); Pt.e2n = F(12); Pt.out = tgt_e;
  hipLaunchKernelGGL(tem_conv_kernel, dim3(4096), dim3(256), 0, stream, Pt);

  // 3) affinity + sigmoid
  hipLaunchKernelGGL(affinity_kernel, dim3(4096), dim3(128), 0, stream,
                     src_e, tgt_e, F(14), F(13), F(16), F(15), (float*)d_out);
}
